// HierarchicalGatedParallelFusion_88768384074397
// MI455X (gfx1250) — matrix operand order repaired and checked by reading, not compiled, not run on hardware
//
#include <hip/hip_runtime.h>
#include <stdint.h>

// Shapes from the reference
#define E_  8
#define B_  16384
#define D_  1024
#define H_  512
#define GH_ 128

typedef __attribute__((ext_vector_type(16))) __bf16 v16bf;
typedef __attribute__((ext_vector_type(8)))  float  v8f;

// POD 16-byte chunks (HIP vector types have user ctors -> not union-safe)
struct U128 { unsigned int w[4]; };
struct F128 { float x[4]; };

// Native conversion: lowers to v_cvt_pk_bf16_f32 pairs (RNE)
__device__ __forceinline__ __bf16 f2bf(float f) { return (__bf16)f; }

// ---------------------------------------------------------------------------
// Weight pre-pass: src fp32 [E][R][C] row-major -> dst bf16 [E][C][R]
// (B-fragment lanes then read 16 contiguous K values = one 32-byte load)
// ---------------------------------------------------------------------------
__global__ void transpose_cvt_bf16(const float* __restrict__ src,
                                   __bf16* __restrict__ dst, int R, int C) {
    __shared__ float tile[16][17];
    const int e = blockIdx.z;
    const float* s = src + (size_t)e * R * C;
    __bf16*      d = dst + (size_t)e * R * C;
    const int c0 = blockIdx.x * 16, r0 = blockIdx.y * 16;
    const int tx = threadIdx.x, ty = threadIdx.y;
    tile[ty][tx] = s[(size_t)(r0 + ty) * C + (c0 + tx)];
    __syncthreads();
    d[(size_t)(c0 + ty) * R + (r0 + tx)] = f2bf(tile[tx][ty]);
}

// ---------------------------------------------------------------------------
// compose pass 1: Wtmp[e] = Wv[e] (fp32, DxH) @ Wvm[e]  -> bf16 row-major [E][D][H]
// ---------------------------------------------------------------------------
__global__ __launch_bounds__(256)
void compose_pass1(const float* __restrict__ Wv,       // [E][D][H]
                   const __bf16* __restrict__ WvmT,    // [E][H][H] n-major
                   __bf16* __restrict__ Wtmp) {        // [E][D][H] row-major
    const int tid = threadIdx.x, lane = tid & 31, wave = tid >> 5;
    const int half = (lane >> 4) & 1, nl = lane & 15, kbA = half * 16;
    const int e = blockIdx.y, row0 = blockIdx.x * 16;

    const float*  ab = Wv   + ((size_t)e * D_ + row0 + nl) * H_;
    const __bf16* bt = WvmT + (size_t)e * H_ * H_;

    v8f acc[4];
    #pragma unroll
    for (int t = 0; t < 4; ++t) acc[t] = {};

    for (int ks = 0; ks < H_ / 32; ++ks) {
        const int k0 = ks * 32;
        union { v16bf v; __bf16 h[16]; } a;
        F128 p0 = *(const F128*)(ab + k0 + kbA);
        F128 p1 = *(const F128*)(ab + k0 + kbA + 4);
        F128 p2 = *(const F128*)(ab + k0 + kbA + 8);
        F128 p3 = *(const F128*)(ab + k0 + kbA + 12);
        #pragma unroll
        for (int j = 0; j < 4; ++j) {
            a.h[j]      = f2bf(p0.x[j]);  a.h[4 + j]  = f2bf(p1.x[j]);
            a.h[8 + j]  = f2bf(p2.x[j]);  a.h[12 + j] = f2bf(p3.x[j]);
        }
        #pragma unroll
        for (int t = 0; t < 4; ++t) {
            const int n = wave * 64 + t * 16 + nl;
            v16bf b = *(const v16bf*)&bt[(size_t)n * H_ + k0 + half * 16];
            acc[t] = __builtin_amdgcn_wmma_f32_16x16x32_bf16(
                false, a.v, false, b, (short)0, acc[t], false, false);
        }
    }
    #pragma unroll
    for (int t = 0; t < 4; ++t) {
        const int n = wave * 64 + t * 16 + nl;
        #pragma unroll
        for (int r = 0; r < 8; ++r)
            Wtmp[((size_t)e * D_ + row0 + r + half * 8) * H_ + n] = f2bf(acc[t][r]);
    }
}

// ---------------------------------------------------------------------------
// compose pass 2: WcT[e] = (Wtmp[e] @ Wo[e])^T  -> bf16 n-major [E][H][D]
// transposed C store is lane-contiguous: one 16B store per tile
// ---------------------------------------------------------------------------
__global__ __launch_bounds__(256)
void compose_pass2(const __bf16* __restrict__ Wtmp,    // [E][D][H] row-major
                   const __bf16* __restrict__ WoT,     // [E][H][H] n-major
                   __bf16* __restrict__ WcT) {         // [E][H][D] n-major
    const int tid = threadIdx.x, lane = tid & 31, wave = tid >> 5;
    const int half = (lane >> 4) & 1, nl = lane & 15, kbA = half * 16;
    const int e = blockIdx.y, row0 = blockIdx.x * 16;

    const __bf16* ab = Wtmp + ((size_t)e * D_ + row0 + nl) * H_;
    const __bf16* bt = WoT  + (size_t)e * H_ * H_;

    v8f acc[4];
    #pragma unroll
    for (int t = 0; t < 4; ++t) acc[t] = {};

    for (int ks = 0; ks < H_ / 32; ++ks) {
        const int k0 = ks * 32;
        union { v16bf v; U128 q[2]; } a;
        a.q[0] = *(const U128*)(ab + k0 + kbA);
        a.q[1] = *(const U128*)(ab + k0 + kbA + 8);
        #pragma unroll
        for (int t = 0; t < 4; ++t) {
            const int n = wave * 64 + t * 16 + nl;
            v16bf b = *(const v16bf*)&bt[(size_t)n * H_ + k0 + half * 16];
            acc[t] = __builtin_amdgcn_wmma_f32_16x16x32_bf16(
                false, a.v, false, b, (short)0, acc[t], false, false);
        }
    }
    #pragma unroll
    for (int t = 0; t < 4; ++t) {
        const int n = wave * 64 + t * 16 + nl;
        union { U128 q; __bf16 h[8]; } o;
        #pragma unroll
        for (int r = 0; r < 8; ++r) o.h[r] = f2bf(acc[t][r]);
        *(U128*)&WcT[((size_t)e * H_ + n) * D_ + row0 + half * 8] = o.q;
    }
}

// ---------------------------------------------------------------------------
// combined bias: bc[e] = bv[e]@Wvm[e]@Wo[e] + bvm[e]@Wo[e] + bo[e]   (fp32)
// ---------------------------------------------------------------------------
__global__ __launch_bounds__(H_)
void bias_compose(const float* __restrict__ bv,  const float* __restrict__ Wvm,
                  const float* __restrict__ bvm, const float* __restrict__ Wo,
                  const float* __restrict__ bo,  float* __restrict__ bc) {
    __shared__ float t[H_];
    const int e = blockIdx.x, tid = threadIdx.x;
    float a = bvm[e * H_ + tid];
    for (int k = 0; k < H_; ++k)
        a = fmaf(bv[e * H_ + k], Wvm[((size_t)e * H_ + k) * H_ + tid], a);
    t[tid] = a;
    __syncthreads();
    float a2 = bo[e * H_ + tid];
    for (int h = 0; h < H_; ++h)
        a2 = fmaf(t[h], Wo[((size_t)e * H_ + h) * H_ + tid], a2);
    bc[e * H_ + tid] = a2;
}

// ---------------------------------------------------------------------------
// gate = softmax(relu(sem @ Wg1 + bg1) @ Wg2 + bg2), stored [B][E] fp32
// ---------------------------------------------------------------------------
__global__ __launch_bounds__(GH_)
void gate_kernel(const float* __restrict__ sem,
                 const float* __restrict__ Wg1, const float* __restrict__ bg1,
                 const float* __restrict__ Wg2, const float* __restrict__ bg2,
                 float* __restrict__ gate) {
    __shared__ float srow[D_];
    __shared__ float hbuf[GH_];
    __shared__ float lg[E_];
    const int b = blockIdx.x, t = threadIdx.x;
    #pragma unroll
    for (int i = 0; i < D_ / GH_; ++i)
        srow[t + i * GH_] = sem[(size_t)b * D_ + t + i * GH_];
    __syncthreads();
    float acc = bg1[t];
    for (int d = 0; d < D_; ++d)
        acc = fmaf(srow[d], Wg1[(size_t)d * GH_ + t], acc);
    hbuf[t] = fmaxf(acc, 0.f);
    __syncthreads();
    if (t < E_) {
        float l = bg2[t];
        for (int j = 0; j < GH_; ++j)
            l = fmaf(hbuf[j], Wg2[(size_t)j * E_ + t], l);
        lg[t] = l;
    }
    __syncthreads();
    if (t < E_) {
        float mx = lg[0];
        #pragma unroll
        for (int k = 1; k < E_; ++k) mx = fmaxf(mx, lg[k]);
        float s = 0.f;
        #pragma unroll
        for (int k = 0; k < E_; ++k) s += __expf(lg[k] - mx);
        gate[(size_t)b * E_ + t] = __expf(lg[t] - mx) / s;
    }
}

// ---------------------------------------------------------------------------
// Main: out = sum_e (gate[:,e] * freq[e]) @ Wc[e]  +  gate @ bc
// 512 threads (16 waves), 64 rows/block; wave = 4 m-tiles x 2 n-tiles.
// Gate folded into A staging; A chunks register-pipelined across barriers;
// B fragments double-buffered over the ks loop (8 WMMAs of load slack).
// ---------------------------------------------------------------------------
#define AP 136   // 128 + 8 bf16 pad
#define NCHUNK (E_ * (D_ / 128))   // 64 staged chunks total

__global__ __launch_bounds__(512)
void moe_fused_gemm(const float* __restrict__ freq,    // [E][B][D] fp32
                    const __bf16* __restrict__ WcT,    // [E][H][D] bf16 n-major
                    const float* __restrict__ bc,      // [E][H] fp32
                    const float* __restrict__ gate,    // [B][E] fp32
                    float* __restrict__ out) {         // [B][H] fp32
    __shared__ __align__(16) __bf16 Abuf[64 * AP];     // 17.4 KB
    __shared__ float gAll[64 * E_];                    // 2 KB

    const int tid  = threadIdx.x;
    const int lane = tid & 31;
    const int wave = tid >> 5;            // 0..15
    const int half = (lane >> 4) & 1;
    const int nl   = lane & 15;
    const int kbA  = half * 16;           // A frag: K = {kbA..+15}
    const int row0 = blockIdx.x * 64;

    // gate tile for these 64 rows: 512 threads load 64*8 values exactly
    gAll[tid] = gate[(size_t)(row0 + (tid >> 3)) * E_ + (tid & 7)];
    __syncthreads();

    // per-thread staging coordinates (loop-invariant)
    int rI[4], cI[4];
    #pragma unroll
    for (int i = 0; i < 4; ++i) {
        const int idx = tid + i * 512;    // over 2048 float4 groups (64 x 32)
        rI[i] = idx >> 5; cI[i] = (idx & 31) * 4;
    }

    v8f acc[4][2];
    #pragma unroll
    for (int mt = 0; mt < 4; ++mt)
        #pragma unroll
        for (int t = 0; t < 2; ++t) acc[mt][t] = {};

    // register double-buffer: preload chunk 0 (e=0, kc=0)
    F128 p[4]; float gs[4];
    #pragma unroll
    for (int i = 0; i < 4; ++i) {
        p[i]  = *(const F128*)&freq[((size_t)row0 + rI[i]) * D_ + cI[i]];
        gs[i] = gAll[rI[i] * E_];
    }

    for (int it = 0; it < NCHUNK; ++it) {
        const int e  = it >> 3;
        const int kc = it & 7;
        const __bf16* wct = WcT + (size_t)e * H_ * D_;

        __syncthreads();                  // Abuf consumers of previous chunk done
        // gate-scaled bf16 stores of the register-buffered chunk
        #pragma unroll
        for (int i = 0; i < 4; ++i) {
            union { unsigned long long u; __bf16 h[4]; } o;
            #pragma unroll
            for (int j = 0; j < 4; ++j) o.h[j] = f2bf(gs[i] * p[i].x[j]);
            *(unsigned long long*)&Abuf[rI[i] * AP + cI[i]] = o.u;
        }
        // issue next chunk's global loads now; latency hides behind the WMMAs
        if (it + 1 < NCHUNK) {
            const int e2 = (it + 1) >> 3, kc2 = (it + 1) & 7;
            #pragma unroll
            for (int i = 0; i < 4; ++i) {
                p[i]  = *(const F128*)&freq[((size_t)e2 * B_ + row0 + rI[i]) * D_
                                            + kc2 * 128 + cI[i]];
                gs[i] = gAll[rI[i] * E_ + e2];
            }
        }
        // preload B fragments for ks=0 (independent of Abuf -> overlaps barrier)
        v16bf bb[2], bn[2];
        #pragma unroll
        for (int t = 0; t < 2; ++t)
            bb[t] = *(const v16bf*)&wct[(size_t)(wave * 32 + t * 16 + nl) * D_
                                        + kc * 128 + half * 16];
        __syncthreads();                  // Abuf ready

        #pragma unroll
        for (int ks = 0; ks < 4; ++ks) {
            // issue B loads for ks+1 before consuming ks (8 WMMAs of slack)
            if (ks < 3) {
                #pragma unroll
                for (int t = 0; t < 2; ++t)
                    bn[t] = *(const v16bf*)&wct[(size_t)(wave * 32 + t * 16 + nl) * D_
                                                + kc * 128 + (ks + 1) * 32 + half * 16];
            }
            union { v16bf v; U128 q[2]; } a[4];
            #pragma unroll
            for (int mt = 0; mt < 4; ++mt) {
                a[mt].q[0] = *(const U128*)&Abuf[(mt * 16 + nl) * AP + ks * 32 + kbA];
                a[mt].q[1] = *(const U128*)&Abuf[(mt * 16 + nl) * AP + ks * 32 + kbA + 8];
            }
            #pragma unroll
            for (int t = 0; t < 2; ++t)
                #pragma unroll
                for (int mt = 0; mt < 4; ++mt)
                    acc[mt][t] = __builtin_amdgcn_wmma_f32_16x16x32_bf16(
                        false, a[mt].v, false, bb[t], (short)0, acc[mt][t], false, false);
            if (ks < 3) {
                #pragma unroll
                for (int t = 0; t < 2; ++t) bb[t] = bn[t];
            }
        }
    }

    // epilogue: + gate-row . bc-column, then store
    float bcv[2][E_];
    #pragma unroll
    for (int t = 0; t < 2; ++t) {
        const int n = wave * 32 + t * 16 + nl;
        #pragma unroll
        for (int e2 = 0; e2 < E_; ++e2) bcv[t][e2] = bc[e2 * H_ + n];
    }
    #pragma unroll
    for (int mt = 0; mt < 4; ++mt) {
        #pragma unroll
        for (int r = 0; r < 8; ++r) {
            const int mrow = mt * 16 + r + half * 8;
            F128 g0 = *(const F128*)&gAll[mrow * E_];
            F128 g1 = *(const F128*)&gAll[mrow * E_ + 4];
            #pragma unroll
            for (int t = 0; t < 2; ++t) {
                const int n = wave * 32 + t * 16 + nl;
                float bias = 0.f;
                #pragma unroll
                for (int j = 0; j < 4; ++j) {
                    bias = fmaf(g0.x[j], bcv[t][j], bias);
                    bias = fmaf(g1.x[j], bcv[t][4 + j], bias);
                }
                out[(size_t)(row0 + mrow) * H_ + n] = acc[mt][t][r] + bias;
            }
        }
    }
}

// ---------------------------------------------------------------------------
extern "C" void kernel_launch(void* const* d_in, const int* in_sizes, int n_in,
                              void* d_out, int out_size, void* d_ws, size_t ws_size,
                              hipStream_t stream) {
    (void)in_sizes; (void)n_in; (void)out_size; (void)ws_size;
    // 0 freq,1 sem,2 Wq,3 bq,4 Wk,5 bk,6 Wv,7 bv,8 Wqm,9 bqm,10 Wkm,11 bkm,
    // 12 Wvm,13 bvm,14 Wo,15 bo,16 Wg1,17 bg1,18 Wg2,19 bg2
    const float* freq = (const float*)d_in[0];
    const float* sem  = (const float*)d_in[1];
    const float* Wv   = (const float*)d_in[6];
    const float* bv   = (const float*)d_in[7];
    const float* Wvm  = (const float*)d_in[12];
    const float* bvm  = (const float*)d_in[13];
    const float* Wo   = (const float*)d_in[14];
    const float* bo   = (const float*)d_in[15];
    const float* Wg1  = (const float*)d_in[16];
    const float* bg1  = (const float*)d_in[17];
    const float* Wg2  = (const float*)d_in[18];
    const float* bg2  = (const float*)d_in[19];
    float* out = (float*)d_out;

    // workspace layout (~25.7 MB)
    char* ws = (char*)d_ws;
    const size_t szHH = (size_t)E_ * H_ * H_ * 2;    // 4.2 MB
    const size_t szDH = (size_t)E_ * D_ * H_ * 2;    // 8.4 MB
    __bf16* WvmT = (__bf16*)(ws);
    __bf16* WoT  = (__bf16*)(ws + szHH);
    __bf16* Wtmp = (__bf16*)(ws + 2 * szHH);
    __bf16* WcT  = (__bf16*)(ws + 2 * szHH + szDH);
    float*  bc   = (float*) (ws + 2 * szHH + 2 * szDH);
    float*  gate = (float*) (ws + 2 * szHH + 2 * szDH + (size_t)E_ * H_ * 4);

    dim3 tb(16, 16, 1);
    transpose_cvt_bf16<<<dim3(H_ / 16, H_ / 16, E_), tb, 0, stream>>>(Wvm, WvmT, H_, H_);
    transpose_cvt_bf16<<<dim3(H_ / 16, H_ / 16, E_), tb, 0, stream>>>(Wo,  WoT,  H_, H_);
    compose_pass1<<<dim3(D_ / 16, E_), 256, 0, stream>>>(Wv, WvmT, Wtmp);
    compose_pass2<<<dim3(D_ / 16, E_), 256, 0, stream>>>(Wtmp, WoT, WcT);
    bias_compose<<<E_, H_, 0, stream>>>(bv, Wvm, bvm, Wo, bo, bc);
    gate_kernel<<<B_, GH_, 0, stream>>>(sem, Wg1, bg1, Wg2, bg2, gate);
    moe_fused_gemm<<<B_ / 64, 512, 0, stream>>>(freq, WcT, bc, gate, out);
}
